// EnhancedWorldGraphEncoder_57483842289957
// MI455X (gfx1250) — matrix operand order repaired and checked by reading, not compiled, not run on hardware
//
#include <hip/hip_runtime.h>

// ---------------------------------------------------------------------------
// Problem constants (match reference)
// ---------------------------------------------------------------------------
#define kN   50000
#define kE   500000
#define kEt  (kE + kN)     // edges + self loops
#define kFIN 64
#define kHID 128
#define kEPS 1e-5f
#define kNEG 0.2f

typedef __bf16 bf16;
typedef __attribute__((ext_vector_type(16))) __bf16 v16bf;
typedef __attribute__((ext_vector_type(8)))  __bf16 v8bf;
typedef __attribute__((ext_vector_type(8)))  float  v8f;

// Monotonic uint encoding of float for atomic max
__device__ __forceinline__ unsigned encf(float f) {
  unsigned u = __float_as_uint(f);
  return (u & 0x80000000u) ? ~u : (u | 0x80000000u);
}
__device__ __forceinline__ float decf(unsigned u) {
  return (u & 0x80000000u) ? __uint_as_float(u & 0x7fffffffu)
                           : __uint_as_float(~u);
}
#define ENC_NEG_INF 0x007FFFFFu   // encf(-inf)

__device__ __forceinline__ float eluf(float x) {
  return x > 0.f ? x : (__expf(x) - 1.f);
}

// ---------------------------------------------------------------------------
// Generic GEMM:  C[nRows,128] (f32) = A[nRows,K] (bf16) @ W[K,128] (f32->bf16)
// 256 threads = 8 waves; each wave computes a 16-row strip over all 8 column
// tiles with V_WMMA_F32_16X16X32_BF16.  W is transposed+converted into LDS.
// ---------------------------------------------------------------------------
#define WT_STRIDE 136   // padded row stride (bf16 elems) to break bank period

__global__ __launch_bounds__(256)
void gemm_bf16_kernel(const bf16* __restrict__ A, const float* __restrict__ W,
                      float* __restrict__ C, int nRows, int K) {
  __shared__ __align__(16) bf16 Wt[kHID * WT_STRIDE];  // Wt[n][k]

  const int tid = threadIdx.x;
  for (int idx = tid; idx < K * kHID; idx += 256) {
    int k = idx >> 7;        // K-major input: W[k][n]
    int n = idx & 127;
    Wt[n * WT_STRIDE + k] = (bf16)W[idx];
  }
  __syncthreads();

  const int wave = tid >> 5;
  const int lane = tid & 31;
  const bool hi  = lane >= 16;
  const int  l15 = lane & 15;

  const int rowA = blockIdx.x * 128 + wave * 16 + l15;   // A-frag row (M=lane&15)
  v8f acc[8];
#pragma unroll
  for (int n = 0; n < 8; ++n)
#pragma unroll
    for (int r = 0; r < 8; ++r) acc[n][r] = 0.f;

  const int kSteps = K >> 5;
  for (int ks = 0; ks < kSteps; ++ks) {
    // A fragment: elems 0..7 = K kb..kb+7, elems 8..15 = K kb+16..kb+23
    const int kbA = ks * 32 + (hi ? 8 : 0);
    v16bf afrag;
    if (rowA < nRows) {
      const bf16* ap = A + rowA * K + kbA;
      v8bf alo = *(const v8bf*)(ap);
      v8bf ahi = *(const v8bf*)(ap + 16);
      afrag = __builtin_shufflevector(alo, ahi, 0,1,2,3,4,5,6,7,
                                                8,9,10,11,12,13,14,15);
    } else {
#pragma unroll
      for (int j = 0; j < 16; ++j) afrag[j] = (bf16)0.f;
    }

    const int kbB = ks * 32 + (hi ? 8 : 0);    // B: same K split as A (kb..kb+7, kb+16..kb+23)
#pragma unroll
    for (int n = 0; n < 8; ++n) {
      const bf16* bp = &Wt[(n * 16 + l15) * WT_STRIDE + kbB];
      v8bf blo = *(const v8bf*)(bp);
      v8bf bhi = *(const v8bf*)(bp + 16);
      v16bf bfrag = __builtin_shufflevector(blo, bhi, 0,1,2,3,4,5,6,7,
                                                      8,9,10,11,12,13,14,15);
      acc[n] = __builtin_amdgcn_wmma_f32_16x16x32_bf16(
          false, afrag, false, bfrag, (short)0, acc[n], false, false);
    }
  }

  // C layout: VGPR r -> row base+r (lanes 0-15) / base+8+r (lanes 16-31), col=lane&15
  const int rowOut = blockIdx.x * 128 + wave * 16 + (hi ? 8 : 0);
#pragma unroll
  for (int n = 0; n < 8; ++n) {
    const int col = n * 16 + l15;
#pragma unroll
    for (int r = 0; r < 8; ++r) {
      int row = rowOut + r;
      if (row < nRows) C[row * kHID + col] = acc[n][r];
    }
  }
}

// ---------------------------------------------------------------------------
// Elementwise / reduction kernels
// ---------------------------------------------------------------------------
__global__ void cvt_bf16_kernel(const float* __restrict__ in,
                                bf16* __restrict__ out, int n) {
  int i = blockIdx.x * blockDim.x + threadIdx.x;
  if (i < n) out[i] = (bf16)in[i];
}

__global__ void fill_f32_kernel(float* p, float v, int n) {
  int i = blockIdx.x * blockDim.x + threadIdx.x;
  if (i < n) p[i] = v;
}
__global__ void fill_u32_kernel(unsigned* p, unsigned v, int n) {
  int i = blockIdx.x * blockDim.x + threadIdx.x;
  if (i < n) p[i] = v;
}

// LayerNorm + ELU over 128 features; one wave per row (wave32, 4 feats/lane)
__global__ void ln_elu_kernel(const float* __restrict__ in,
                              const float* __restrict__ g,
                              const float* __restrict__ b,
                              float* __restrict__ h, bf16* __restrict__ hB,
                              int n) {
  int wave = (blockIdx.x * blockDim.x + threadIdx.x) >> 5;
  int lane = threadIdx.x & 31;
  if (wave >= n) return;
  float v[4]; float s = 0.f;
#pragma unroll
  for (int j = 0; j < 4; ++j) { v[j] = in[wave * kHID + lane * 4 + j]; s += v[j]; }
#pragma unroll
  for (int off = 16; off; off >>= 1) s += __shfl_xor(s, off);
  float mu = s * (1.f / kHID);
  float q = 0.f;
#pragma unroll
  for (int j = 0; j < 4; ++j) { float d = v[j] - mu; q += d * d; }
#pragma unroll
  for (int off = 16; off; off >>= 1) q += __shfl_xor(q, off);
  float r = rsqrtf(q * (1.f / kHID) + kEPS);
#pragma unroll
  for (int j = 0; j < 4; ++j) {
    int f = lane * 4 + j;
    float y = (v[j] - mu) * r * g[f] + b[f];
    y = eluf(y);
    int idx = wave * kHID + f;
    h[idx] = y;
    hB[idx] = (bf16)y;
  }
}

// a_src[i] = hw[i] . att_src ; a_dst[i] = hw[i] . att_dst   (wave per row)
__global__ void attdot_kernel(const float* __restrict__ hw,
                              const float* __restrict__ att_src,
                              const float* __restrict__ att_dst,
                              float* __restrict__ a_s, float* __restrict__ a_d,
                              int n) {
  int wave = (blockIdx.x * blockDim.x + threadIdx.x) >> 5;
  int lane = threadIdx.x & 31;
  if (wave >= n) return;
  float s1 = 0.f, s2 = 0.f;
#pragma unroll
  for (int j = 0; j < 4; ++j) {
    int f = lane * 4 + j;
    float v = hw[wave * kHID + f];
    s1 += v * att_src[f];
    s2 += v * att_dst[f];
  }
#pragma unroll
  for (int off = 16; off; off >>= 1) { s1 += __shfl_xor(s1, off); s2 += __shfl_xor(s2, off); }
  if (!lane) { a_s[wave] = s1; a_d[wave] = s2; }
}

// Edge pass 1: alpha = leaky_relu(a_src[s] + a_dst[d]); segment max into amax
__global__ void edge_max_kernel(const int* __restrict__ src,
                                const int* __restrict__ dst,
                                const float* __restrict__ a_s,
                                const float* __restrict__ a_d,
                                float* __restrict__ alpha,
                                unsigned* __restrict__ amax) {
  int e = blockIdx.x * blockDim.x + threadIdx.x;
  if (e >= kEt) return;
  int s = (e < kE) ? src[e] : (e - kE);
  int d = (e < kE) ? dst[e] : (e - kE);
  float a = a_s[s] + a_d[d];
  a = a > 0.f ? a : kNEG * a;
  alpha[e] = a;
  atomicMax(&amax[d], encf(a));
}

// Edge pass 2: ex = exp(alpha - amax[d]); denom[d] += ex; alpha <- ex
__global__ void edge_exp_kernel(const int* __restrict__ dst,
                                float* __restrict__ alpha,
                                const unsigned* __restrict__ amax,
                                float* __restrict__ denom) {
  int e = blockIdx.x * blockDim.x + threadIdx.x;
  if (e >= kEt) return;
  int d = (e < kE) ? dst[e] : (e - kE);
  float ex = __expf(alpha[e] - decf(amax[d]));
  alpha[e] = ex;
  atomicAdd(&denom[d], ex);
}

// Edge pass 3 (wave per edge): agg[d] += hw[s] * (ex / denom[d])
__global__ void edge_scatter_kernel(const int* __restrict__ src,
                                    const int* __restrict__ dst,
                                    const float* __restrict__ alpha,
                                    const float* __restrict__ denom,
                                    const float* __restrict__ hw,
                                    float* __restrict__ agg) {
  int e    = (blockIdx.x * blockDim.x + threadIdx.x) >> 5;
  int lane = threadIdx.x & 31;
  if (e >= kEt) return;
  int s = (e < kE) ? src[e] : (e - kE);
  int d = (e < kE) ? dst[e] : (e - kE);
  float w = alpha[e] / denom[d];
  const float* hs = hw + s * kHID;
  float* ag = agg + d * kHID;
#pragma unroll
  for (int j = 0; j < 4; ++j) {
    int f = lane * 4 + j;
    atomicAdd(&ag[f], hs[f] * w);
  }
}

// BatchNorm stats over node axis (feature-per-thread, 128 threads/block)
__global__ void bnstats_kernel(const float* __restrict__ agg,
                               const float* __restrict__ bias,
                               float* __restrict__ bnsum, float* __restrict__ bnsq,
                               int n) {
  int f = threadIdx.x;
  float bv = bias[f];
  float s = 0.f, q = 0.f;
  for (int i = blockIdx.x; i < n; i += gridDim.x) {
    float u = agg[i * kHID + f] + bv;
    s += u; q += u * u;
  }
  atomicAdd(&bnsum[f], s);
  atomicAdd(&bnsq[f], q);
}

// BN apply + ELU + residual; refresh h (f32) and hB (bf16)
__global__ void bnapply_kernel(const float* __restrict__ agg,
                               const float* __restrict__ bias,
                               const float* __restrict__ bnsum,
                               const float* __restrict__ bnsq,
                               const float* __restrict__ g,
                               const float* __restrict__ b,
                               float* __restrict__ h, bf16* __restrict__ hB,
                               int total) {
  int idx = blockIdx.x * blockDim.x + threadIdx.x;
  if (idx >= total) return;
  int f = idx & (kHID - 1);
  float mu  = bnsum[f] * (1.f / kN);
  float var = bnsq[f] * (1.f / kN) - mu * mu;
  float u = agg[idx] + bias[f];
  float y = (u - mu) * rsqrtf(var + kEPS) * g[f] + b[f];
  y = eluf(y);
  float hn = y + h[idx];
  h[idx]  = hn;
  hB[idx] = (bf16)hn;
}

// Attention score: s[i] = relu(hw[i]+b1) . W2 + b2 ; track global max
__global__ void score_kernel(const float* __restrict__ hw,
                             const float* __restrict__ b1,
                             const float* __restrict__ W2,
                             const float* __restrict__ b2,
                             float* __restrict__ s_raw, unsigned* __restrict__ smax,
                             int n) {
  int wave = (blockIdx.x * blockDim.x + threadIdx.x) >> 5;
  int lane = threadIdx.x & 31;
  if (wave >= n) return;
  float s = 0.f;
#pragma unroll
  for (int j = 0; j < 4; ++j) {
    int f = lane * 4 + j;
    float t = hw[wave * kHID + f] + b1[f];
    s += fmaxf(t, 0.f) * W2[f];
  }
#pragma unroll
  for (int off = 16; off; off >>= 1) s += __shfl_xor(s, off);
  if (!lane) {
    float sv = s + b2[0];
    s_raw[wave] = sv;
    atomicMax(smax, encf(sv));
  }
}

// Mean / max / attention pooling accumulators (feature-per-thread)
__global__ void pool_kernel(const float* __restrict__ h,
                            const float* __restrict__ s_raw,
                            const unsigned* __restrict__ smax,
                            float* __restrict__ meanSum,
                            unsigned* __restrict__ maxEnc,
                            float* __restrict__ attnSum,
                            float* __restrict__ sumexp, int n) {
  int f = threadIdx.x;
  float sm = decf(*smax);
  float ms = 0.f, mx = -3.4e38f, as = 0.f, se = 0.f;
  for (int i = blockIdx.x; i < n; i += gridDim.x) {
    float hv = h[i * kHID + f];
    ms += hv;
    mx = fmaxf(mx, hv);
    float e = __expf(s_raw[i] - sm);
    as += hv * e;
    if (f == 0) se += e;
  }
  atomicAdd(&meanSum[f], ms);
  atomicMax(&maxEnc[f], encf(mx));
  atomicAdd(&attnSum[f], as);
  if (f == 0) atomicAdd(sumexp, se);
}

// Final fusion MLP + LayerNorm; single block of 128 threads
__global__ void final_kernel(const float* __restrict__ meanSum,
                             const unsigned* __restrict__ maxEnc,
                             const float* __restrict__ attnSum,
                             const float* __restrict__ sumexp,
                             const float* __restrict__ mean_W, const float* __restrict__ mean_b,
                             const float* __restrict__ max_W,  const float* __restrict__ max_b,
                             const float* __restrict__ fus_W1, const float* __restrict__ fus_b1,
                             const float* __restrict__ fus_W2, const float* __restrict__ fus_b2,
                             const float* __restrict__ ln_g,   const float* __restrict__ ln_b,
                             float* __restrict__ out) {
  __shared__ float mv[128], xv[128], av[128], comb[384], g1[128], red[128];
  int t = threadIdx.x;
  mv[t] = meanSum[t] * (1.f / kN);
  xv[t] = decf(maxEnc[t]);
  av[t] = attnSum[t] / sumexp[0];
  __syncthreads();

  float m = mean_b[t], x = max_b[t], a = 0.f;
  for (int k = 0; k < 128; ++k) {
    m += mv[k] * mean_W[k * 128 + t];
    x += xv[k] * max_W[k * 128 + t];
    a += av[k] * mean_W[k * 128 + t];   // attn pool shares mean_W, no bias
  }
  comb[t] = m; comb[128 + t] = x; comb[256 + t] = a;
  __syncthreads();

  float gacc = fus_b1[t];
  for (int k = 0; k < 384; ++k) gacc += comb[k] * fus_W1[k * 128 + t];
  g1[t] = fmaxf(gacc, 0.f);
  __syncthreads();

  float g2v = fus_b2[t];
  for (int k = 0; k < 128; ++k) g2v += g1[k] * fus_W2[k * 128 + t];

  red[t] = g2v; __syncthreads();
  for (int s2 = 64; s2; s2 >>= 1) { if (t < s2) red[t] += red[t + s2]; __syncthreads(); }
  float mu = red[0] * (1.f / 128.f); __syncthreads();
  red[t] = (g2v - mu) * (g2v - mu); __syncthreads();
  for (int s2 = 64; s2; s2 >>= 1) { if (t < s2) red[t] += red[t + s2]; __syncthreads(); }
  float var = red[0] * (1.f / 128.f);
  out[t] = (g2v - mu) * rsqrtf(var + kEPS) * ln_g[t] + ln_b[t];
}

// ---------------------------------------------------------------------------
// Host launcher
// ---------------------------------------------------------------------------
extern "C" void kernel_launch(void* const* d_in, const int* in_sizes, int n_in,
                              void* d_out, int out_size, void* d_ws, size_t ws_size,
                              hipStream_t stream) {
  (void)in_sizes; (void)n_in; (void)out_size; (void)ws_size;

  // ---- inputs (flattened dict order) ----
  const float* x    = (const float*)d_in[0];
  const int*   ei   = (const int*)d_in[1];     // [2, E]
  const float* npW  = (const float*)d_in[2];
  const float* npb  = (const float*)d_in[3];
  const float* ln0g = (const float*)d_in[4];
  const float* ln0b = (const float*)d_in[5];
  // gat layer L: indices 6+6L .. 11+6L : W, att_src, att_dst, bias, bn_g, bn_b
  const float* attnW1 = (const float*)d_in[24];
  const float* attnb1 = (const float*)d_in[25];
  const float* attnW2 = (const float*)d_in[26];
  const float* attnb2 = (const float*)d_in[27];
  const float* meanW  = (const float*)d_in[28];
  const float* meanb  = (const float*)d_in[29];
  const float* maxW   = (const float*)d_in[30];
  const float* maxb   = (const float*)d_in[31];
  const float* fusW1  = (const float*)d_in[32];
  const float* fusb1  = (const float*)d_in[33];
  const float* fusW2  = (const float*)d_in[34];
  const float* fusb2  = (const float*)d_in[35];
  const float* flng   = (const float*)d_in[36];
  const float* flnb   = (const float*)d_in[37];

  const int* src = ei;
  const int* dst = ei + kE;

  // ---- workspace carve ----
  char* w = (char*)d_ws;
  bf16* xB = (bf16*)w;               w += (size_t)kN * kFIN * 2;
  bf16* hB = (bf16*)w;               w += (size_t)kN * kHID * 2;
  float* h   = (float*)w;            w += (size_t)kN * kHID * 4;
  float* hw  = (float*)w;            w += (size_t)kN * kHID * 4;
  float* agg = (float*)w;            w += (size_t)kN * kHID * 4;
  float* a_s = (float*)w;            w += (size_t)kN * 4;
  float* a_d = (float*)w;            w += (size_t)kN * 4;
  float* alpha = (float*)w;          w += (size_t)kEt * 4;
  unsigned* amax = (unsigned*)w;     w += (size_t)kN * 4;
  float* denom = (float*)w;          w += (size_t)kN * 4;
  float* s_raw = (float*)w;          w += (size_t)kN * 4;
  float* bnsum = (float*)w;          w += 128 * 4;
  float* bnsq  = (float*)w;          w += 128 * 4;
  unsigned* smax = (unsigned*)w;     w += 4;
  float* sumexp  = (float*)w;        w += 4;
  float* meanSum = (float*)w;        w += 128 * 4;
  unsigned* maxEnc = (unsigned*)w;   w += 128 * 4;
  float* attnSum = (float*)w;        w += 128 * 4;

  const int gemmGrid = (kN + 127) / 128;
  const int rowWaveBlocks = (kN * 32 + 255) / 256;
  const int edgeBlocks = (kEt + 255) / 256;
  const int edgeWaveBlocks = (kEt * 32 + 255) / 256;

  // 1) x -> bf16
  cvt_bf16_kernel<<<(kN * kFIN + 255) / 256, 256, 0, stream>>>(x, xB, kN * kFIN);

  // 2) node projection GEMM (K=64) -> hw
  gemm_bf16_kernel<<<gemmGrid, 256, 0, stream>>>(xB, npW, hw, kN, kFIN);

  // 3) hw + bias handled in LN? Reference: LN(x@W + b). Add bias inside LN input:
  //    fold by pre-adding bias via a tiny kernel: reuse bnapply-style path is
  //    overkill; LN is shift-invariant in mean but scale matters, so add bias first.
  //    (LN subtracts the row mean, but bias changes variance -> must add.)
  {
    // add bias elementwise into hw
    // reuse fill grid style
    struct Local {};
    // simple lambda-free kernel below
  }
  // bias add kernel (defined inline below via a dedicated kernel)
  extern __global__ void bias_add_kernel(float*, const float*, int);
  bias_add_kernel<<<(kN * kHID + 255) / 256, 256, 0, stream>>>(hw, npb, kN * kHID);

  // 4) LN + ELU -> h, hB
  ln_elu_kernel<<<rowWaveBlocks, 256, 0, stream>>>(hw, ln0g, ln0b, h, hB, kN);

  // 5) GAT layers
  for (int L = 0; L < 3; ++L) {
    const float* Wl   = (const float*)d_in[6 + 6 * L + 0];
    const float* asrc = (const float*)d_in[6 + 6 * L + 1];
    const float* adst = (const float*)d_in[6 + 6 * L + 2];
    const float* bl   = (const float*)d_in[6 + 6 * L + 3];
    const float* bng  = (const float*)d_in[6 + 6 * L + 4];
    const float* bnb  = (const float*)d_in[6 + 6 * L + 5];

    fill_u32_kernel<<<(kN + 255) / 256, 256, 0, stream>>>(amax, ENC_NEG_INF, kN);
    fill_f32_kernel<<<(kN + 255) / 256, 256, 0, stream>>>(denom, 0.f, kN);
    fill_f32_kernel<<<(kN * kHID + 255) / 256, 256, 0, stream>>>(agg, 0.f, kN * kHID);
    fill_f32_kernel<<<1, 256, 0, stream>>>(bnsum, 0.f, 128);
    fill_f32_kernel<<<1, 256, 0, stream>>>(bnsq, 0.f, 128);

    gemm_bf16_kernel<<<gemmGrid, 256, 0, stream>>>(hB, Wl, hw, kN, kHID);
    attdot_kernel<<<rowWaveBlocks, 256, 0, stream>>>(hw, asrc, adst, a_s, a_d, kN);
    edge_max_kernel<<<edgeBlocks, 256, 0, stream>>>(src, dst, a_s, a_d, alpha, amax);
    edge_exp_kernel<<<edgeBlocks, 256, 0, stream>>>(dst, alpha, amax, denom);
    edge_scatter_kernel<<<edgeWaveBlocks, 256, 0, stream>>>(src, dst, alpha, denom, hw, agg);
    bnstats_kernel<<<256, 128, 0, stream>>>(agg, bl, bnsum, bnsq, kN);
    bnapply_kernel<<<(kN * kHID + 255) / 256, 256, 0, stream>>>(
        agg, bl, bnsum, bnsq, bng, bnb, h, hB, kN * kHID);
  }

  // 6) attention scores: GEMM h@attn_W1 -> hw, then score + softmax max
  gemm_bf16_kernel<<<gemmGrid, 256, 0, stream>>>(hB, attnW1, hw, kN, kHID);
  fill_u32_kernel<<<1, 256, 0, stream>>>(smax, ENC_NEG_INF, 1);
  fill_f32_kernel<<<1, 256, 0, stream>>>(sumexp, 0.f, 1);
  fill_f32_kernel<<<1, 256, 0, stream>>>(meanSum, 0.f, 128);
  fill_u32_kernel<<<1, 256, 0, stream>>>(maxEnc, ENC_NEG_INF, 128);
  fill_f32_kernel<<<1, 256, 0, stream>>>(attnSum, 0.f, 128);

  score_kernel<<<rowWaveBlocks, 256, 0, stream>>>(hw, attnb1, attnW2, attnb2,
                                                  s_raw, smax, kN);
  pool_kernel<<<128, 128, 0, stream>>>(h, s_raw, smax, meanSum, maxEnc,
                                       attnSum, sumexp, kN);
  final_kernel<<<1, 128, 0, stream>>>(meanSum, maxEnc, attnSum, sumexp,
                                      meanW, meanb, maxW, maxb,
                                      fusW1, fusb1, fusW2, fusb2,
                                      flng, flnb, (float*)d_out);
}

// bias add (declared above, defined here)
__global__ void bias_add_kernel(float* __restrict__ p,
                                const float* __restrict__ b, int total) {
  int i = blockIdx.x * blockDim.x + threadIdx.x;
  if (i < total) p[i] += b[i & (kHID - 1)];
}
